// OptimizedSequenceEncoder_67070209294990
// MI455X (gfx1250) — matrix operand order repaired and checked by reading, not compiled, not run on hardware
//
#include <hip/hip_runtime.h>
#include <hip/hip_bf16.h>

typedef __bf16 bf16_t;
typedef __attribute__((ext_vector_type(16))) __bf16 v16bf;
typedef __attribute__((ext_vector_type(8)))  float  v8f;

#define GEMM_STRIDE 40   // 32 k-halfs + 8 pad -> rows 16B-aligned, 20-bank lane skew
#define AT_STRIDE   72   // 64 k-halfs + 8 pad -> rows 16B-aligned

__device__ __forceinline__ unsigned pack_bf16(float a, float b) {
  union { bf16_t h[2]; unsigned u; } p;
  p.h[0] = (bf16_t)a; p.h[1] = (bf16_t)b;
  return p.u;
}

// A fragment (16x32 bf16, MxK) from LDS tile stored [m][k] (k contiguous).
// ISA 7.12.2: lane m = lane&15, grp = lane>>4. vgpr j holds k = 2j + (j>=4?8:0) + 8*grp.
// Per lane that is two contiguous 16B runs: halfs [grp*8, +8) and [grp*8+16, +8).
__device__ __forceinline__ v16bf gather_a(const bf16_t* lds, int rowBase, int stride, int kOff) {
  const int lane = threadIdx.x & 31;
  const int grp  = lane >> 4;
  const int m    = lane & 15;
  const bf16_t* p = lds + (rowBase + m) * stride + grp * 8 + kOff;
  union { uint4 q[2]; v16bf v; } f;
  f.q[0] = *(const uint4*)(p);        // j=0..3
  f.q[1] = *(const uint4*)(p + 16);   // j=4..7
  return f.v;
}

// B fragment (32x16 bf16, KxN) from LDS tile stored [n][k] (k contiguous).
// lane n = (lane&15)+nBase, same k order as A: halfs [grp*8, +8) and [grp*8+16, +8).
__device__ __forceinline__ v16bf gather_b(const bf16_t* lds, int nBase, int stride, int kOff) {
  const int lane = threadIdx.x & 31;
  const int grp  = lane >> 4;
  const int n    = (lane & 15) + nBase;
  const bf16_t* p = lds + n * stride + grp * 8 + kOff;
  union { uint4 q[2]; v16bf v; } f;
  f.q[0] = *(const uint4*)(p);
  f.q[1] = *(const uint4*)(p + 16);
  return f.v;
}

__device__ __forceinline__ v8f wmma_bf16(v16bf a, v16bf b, v8f c) {
  return __builtin_amdgcn_wmma_f32_16x16x32_bf16(false, a, false, b, (short)0, c, false, false);
}

// ---------------------------------------------------------------------------
// C[M,N] = A[M,K] @ W[K,N] + bias.  f32 in/out, bf16 WMMA compute.
// Block tile 128x256, BK=32, 256 threads = 8 waves in 2(M)x4(N) grid.
// Each wave: 4 A-frags x 4 B-frags = 16 WMMAs/k-step (1 ds_load_b128 per WMMA).
// ---------------------------------------------------------------------------
__global__ __launch_bounds__(256)
void gemm_bias_kernel(const float* __restrict__ A, const float* __restrict__ W,
                      const float* __restrict__ bias, float* __restrict__ C,
                      int M, int N, int K) {
  __shared__ bf16_t As[128 * GEMM_STRIDE];  // [m][k]
  __shared__ bf16_t Bs[256 * GEMM_STRIDE];  // [n][k] (transposed at load)
  const int tid    = threadIdx.x;
  const int wave   = tid >> 5;
  const int waveM  = wave >> 2;             // 0..1 -> 64-row group
  const int waveN  = wave & 3;              // 0..3 -> 64-col group
  const int lane   = tid & 31;
  const int grp    = lane >> 4;
  const int nl     = lane & 15;
  const int blockM = blockIdx.y * 128;
  const int blockN = blockIdx.x * 256;

  v8f acc[4][4] = {};

  for (int kb = 0; kb < K; kb += 32) {
    // Stage A tile 128x32 (f32 -> bf16)
#pragma unroll
    for (int i = 0; i < 4; ++i) {
      const int v = tid + i * 256;            // 1024 float4
      const int row = v >> 3, c4 = (v & 7) * 4;
      const float4 d = *(const float4*)(A + (size_t)(blockM + row) * K + kb + c4);
      unsigned* dst = (unsigned*)(As + row * GEMM_STRIDE + c4);
      dst[0] = pack_bf16(d.x, d.y);
      dst[1] = pack_bf16(d.z, d.w);
    }
    // Stage W tile 32x256 transposed -> Bs[n][k]
#pragma unroll
    for (int i = 0; i < 8; ++i) {
      const int v = tid + i * 256;            // 2048 float4
      const int kr = v >> 6, c4 = (v & 63) * 4;
      const float4 d = *(const float4*)(W + (size_t)(kb + kr) * N + blockN + c4);
      Bs[(c4 + 0) * GEMM_STRIDE + kr] = (bf16_t)d.x;
      Bs[(c4 + 1) * GEMM_STRIDE + kr] = (bf16_t)d.y;
      Bs[(c4 + 2) * GEMM_STRIDE + kr] = (bf16_t)d.z;
      Bs[(c4 + 3) * GEMM_STRIDE + kr] = (bf16_t)d.w;
    }
    if (kb + 32 < K) {  // -> global_prefetch_b8
      __builtin_prefetch(A + (size_t)(blockM + (tid >> 1)) * K + kb + 32, 0, 1);
      __builtin_prefetch(W + (size_t)(kb + 32 + (tid >> 3)) * N + blockN + (tid & 7) * 32, 0, 1);
    }
    __syncthreads();

    v16bf a[4];
#pragma unroll
    for (int mt = 0; mt < 4; ++mt)
      a[mt] = gather_a(As, waveM * 64 + mt * 16, GEMM_STRIDE, 0);
#pragma unroll
    for (int nt = 0; nt < 4; ++nt) {
      const v16bf b = gather_b(Bs, waveN * 64 + nt * 16, GEMM_STRIDE, 0);
#pragma unroll
      for (int mt = 0; mt < 4; ++mt)
        acc[mt][nt] = wmma_bf16(a[mt], b, acc[mt][nt]);
    }
    __syncthreads();
  }

#pragma unroll
  for (int nt = 0; nt < 4; ++nt) {
    const int col = blockN + waveN * 64 + nt * 16 + nl;
    const float bv = bias ? bias[col] : 0.0f;
#pragma unroll
    for (int mt = 0; mt < 4; ++mt) {
#pragma unroll
      for (int r = 0; r < 8; ++r) {
        const int row = blockM + waveM * 64 + mt * 16 + r + grp * 8; // C: M = r + 8*grp
        C[(size_t)row * N + col] = acc[mt][nt][r] + bv;
      }
    }
  }
}

// ---------------------------------------------------------------------------
// Flash attention: grid (B*NH=128, S/128=8), 256 threads = 8 waves.
// Wave owns 16 query rows. 16 kv tiles of 64 keys. HD=64, SCALE=1/8.
// qkv layout: [(b*S+s)*3H + which*H + head*64 + hd]
// ---------------------------------------------------------------------------
__global__ __launch_bounds__(256)
void attention_kernel(const float* __restrict__ qkv, float* __restrict__ out) {
  __shared__ bf16_t Ps[128 * AT_STRIDE];  // Q staging, then P (probabilities)
  __shared__ bf16_t Ks[64 * AT_STRIDE];   // [key][hd]
  __shared__ bf16_t Vt[64 * AT_STRIDE];   // [hd][key]
  const int tid  = threadIdx.x;
  const int wave = tid >> 5;
  const int lane = tid & 31;
  const int grp  = lane >> 4;
  const int nl   = lane & 15;
  const int bh   = blockIdx.x;
  const int b    = bh >> 4, head = bh & 15;
  const int qbase = blockIdx.y * 128;
  const size_t base = (size_t)b * 1024 * 3072 + head * 64;
  const float CS = 0.125f * 1.44269504088896f;  // SCALE * log2(e)

  // Stage Q tile 128x64 -> bf16
#pragma unroll
  for (int i = 0; i < 8; ++i) {
    const int v = tid + i * 256;               // 2048 float4
    const int row = v >> 4, c4 = (v & 15) * 4;
    const float4 d = *(const float4*)(qkv + base + (size_t)(qbase + row) * 3072 + c4);
    unsigned* dst = (unsigned*)(Ps + row * AT_STRIDE + c4);
    dst[0] = pack_bf16(d.x, d.y);
    dst[1] = pack_bf16(d.z, d.w);
  }
  __syncthreads();
  // Preload Q fragments (hd = K dim, two 32-slices); Ps is then reused for P.
  const v16bf aq0 = gather_a(Ps, wave * 16, AT_STRIDE, 0);
  const v16bf aq1 = gather_a(Ps, wave * 16, AT_STRIDE, 32);

  float m_i[8], l_i[8];
#pragma unroll
  for (int r = 0; r < 8; ++r) { m_i[r] = -3.0e38f; l_i[r] = 0.0f; }
  v8f accO[4] = {};

  for (int kt = 0; kt < 16; ++kt) {
    // Stage K tile [key][hd] and V tile transposed [hd][key]
#pragma unroll
    for (int i = 0; i < 4; ++i) {
      const int v = tid + i * 256;             // 1024 float4
      const int row = v >> 4, c4 = (v & 15) * 4;
      const size_t srow = (size_t)(kt * 64 + row) * 3072;
      const float4 dk = *(const float4*)(qkv + base + 1024 + srow + c4);
      unsigned* dst = (unsigned*)(Ks + row * AT_STRIDE + c4);
      dst[0] = pack_bf16(dk.x, dk.y);
      dst[1] = pack_bf16(dk.z, dk.w);
      const float4 dv = *(const float4*)(qkv + base + 2048 + srow + c4);
      Vt[(c4 + 0) * AT_STRIDE + row] = (bf16_t)dv.x;
      Vt[(c4 + 1) * AT_STRIDE + row] = (bf16_t)dv.y;
      Vt[(c4 + 2) * AT_STRIDE + row] = (bf16_t)dv.z;
      Vt[(c4 + 3) * AT_STRIDE + row] = (bf16_t)dv.w;
    }
    __syncthreads();

    // S = Q @ K^T : B fragment n=key, k=hd from Ks[n][k]
    v8f accS[4] = {};
#pragma unroll
    for (int nt = 0; nt < 4; ++nt) {
      const v16bf b0 = gather_b(Ks, nt * 16, AT_STRIDE, 0);
      accS[nt] = wmma_bf16(aq0, b0, accS[nt]);
      const v16bf b1 = gather_b(Ks, nt * 16, AT_STRIDE, 32);
      accS[nt] = wmma_bf16(aq1, b1, accS[nt]);
    }

    // Online softmax. Row m = r + 8*grp lives in 16 lanes of one half-wave.
#pragma unroll
    for (int r = 0; r < 8; ++r) {
      float mx = fmaxf(fmaxf(accS[0][r], accS[1][r]), fmaxf(accS[2][r], accS[3][r]));
      mx = fmaxf(mx, __shfl_xor(mx, 1));
      mx = fmaxf(mx, __shfl_xor(mx, 2));
      mx = fmaxf(mx, __shfl_xor(mx, 4));
      mx = fmaxf(mx, __shfl_xor(mx, 8));
      const float mnew = fmaxf(m_i[r], mx);
      const float al = exp2f((m_i[r] - mnew) * CS);
      float rs = 0.0f;
#pragma unroll
      for (int nt = 0; nt < 4; ++nt) {
        const float p = exp2f((accS[nt][r] - mnew) * CS);
        accS[nt][r] = p;                 // reuse accS as probabilities
        rs += p;
      }
      rs += __shfl_xor(rs, 1); rs += __shfl_xor(rs, 2);
      rs += __shfl_xor(rs, 4); rs += __shfl_xor(rs, 8);
      l_i[r] = l_i[r] * al + rs;
      m_i[r] = mnew;
#pragma unroll
      for (int nt = 0; nt < 4; ++nt) accO[nt][r] *= al;
    }

    // Write P (bf16) into this wave's own 16 rows of Ps, regather as A fragment.
#pragma unroll
    for (int nt = 0; nt < 4; ++nt)
#pragma unroll
      for (int r = 0; r < 8; ++r)
        Ps[(wave * 16 + r + grp * 8) * AT_STRIDE + nt * 16 + nl] = (bf16_t)accS[nt][r];

    // O += P @ V : B fragment n=hd, k=key from Vt[n][k]
#pragma unroll
    for (int kb = 0; kb < 2; ++kb) {
      const v16bf ap = gather_a(Ps, wave * 16, AT_STRIDE, kb * 32);
#pragma unroll
      for (int nt = 0; nt < 4; ++nt) {
        const v16bf bv = gather_b(Vt, nt * 16, AT_STRIDE, kb * 32);
        accO[nt] = wmma_bf16(ap, bv, accO[nt]);
      }
    }
    __syncthreads();
  }

  // Normalize and write O back to [b, s, head*64+hd] (f32)
#pragma unroll
  for (int nt = 0; nt < 4; ++nt) {
#pragma unroll
    for (int r = 0; r < 8; ++r) {
      const int row = qbase + wave * 16 + r + grp * 8;
      const int hd  = nt * 16 + nl;
      out[((size_t)b * 1024 + row) * 1024 + head * 64 + hd] = accO[nt][r] / l_i[r];
    }
  }
}

// ---------------------------------------------------------------------------
// LayerNorm (+ optional residual). One block per row of `ncols`==1024.
// ---------------------------------------------------------------------------
__global__ __launch_bounds__(256)
void ln_kernel(const float* __restrict__ x, const float* __restrict__ res,
               const float* __restrict__ g, const float* __restrict__ be,
               float* __restrict__ out, int ncols) {
  __shared__ float redA[8], redB[8];
  const int row = blockIdx.x;
  const int tid = threadIdx.x;
  const int lane = tid & 31, wave = tid >> 5;
  float v[4];
  float s = 0.0f, sq = 0.0f;
#pragma unroll
  for (int i = 0; i < 4; ++i) {
    const int c = tid + i * 256;
    float xv = x[(size_t)row * ncols + c];
    if (res) xv += res[(size_t)row * ncols + c];
    v[i] = xv; s += xv; sq += xv * xv;
  }
#pragma unroll
  for (int m = 16; m >= 1; m >>= 1) { s += __shfl_xor(s, m); sq += __shfl_xor(sq, m); }
  if (lane == 0) { redA[wave] = s; redB[wave] = sq; }
  __syncthreads();
  if (tid == 0) {
    float ts = 0.0f, tq = 0.0f;
    for (int w = 0; w < 8; ++w) { ts += redA[w]; tq += redB[w]; }
    redA[0] = ts; redB[0] = tq;
  }
  __syncthreads();
  const float mean = redA[0] / (float)ncols;
  const float var  = redB[0] / (float)ncols - mean * mean;
  const float inv  = rsqrtf(var + 1e-5f);
#pragma unroll
  for (int i = 0; i < 4; ++i) {
    const int c = tid + i * 256;
    out[(size_t)row * ncols + c] = (v[i] - mean) * inv * g[c] + be[c];
  }
}

// pooled[b, c] = mean over S of h[b, s, c]
__global__ __launch_bounds__(256)
void pool_kernel(const float* __restrict__ h, float* __restrict__ pooled) {
  const int idx = blockIdx.x * 256 + threadIdx.x;   // 8192 = B*H
  const int b = idx >> 10, c = idx & 1023;
  float s = 0.0f;
  for (int t = 0; t < 1024; ++t) s += h[((size_t)b * 1024 + t) * 1024 + c];
  pooled[idx] = s * (1.0f / 1024.0f);
}

// 8x1024 @ 1024x1024 + bias (tiny, f32 scalar path)
__global__ __launch_bounds__(256)
void smallgemm_kernel(const float* __restrict__ A, const float* __restrict__ W,
                      const float* __restrict__ bias, float* __restrict__ C) {
  const int idx = blockIdx.x * 256 + threadIdx.x;   // 8192 outputs
  const int b = idx >> 10, n = idx & 1023;
  float s = bias[n];
  for (int k = 0; k < 1024; ++k) s = fmaf(A[b * 1024 + k], W[(size_t)k * 1024 + n], s);
  C[idx] = s;
}

extern "C" void kernel_launch(void* const* d_in, const int* in_sizes, int n_in,
                              void* d_out, int out_size, void* d_ws, size_t ws_size,
                              hipStream_t stream) {
  (void)in_sizes; (void)n_in; (void)out_size; (void)ws_size;
  const float* x     = (const float*)d_in[0];
  const float* W_in  = (const float*)d_in[1];
  const float* b_in  = (const float*)d_in[2];
  const float* W_qkv = (const float*)d_in[3];
  const float* b_qkv = (const float*)d_in[4];
  const float* W_o   = (const float*)d_in[5];
  const float* b_o   = (const float*)d_in[6];
  const float* g1    = (const float*)d_in[7];
  const float* be1   = (const float*)d_in[8];
  const float* W_out = (const float*)d_in[9];
  const float* b_out = (const float*)d_in[10];
  const float* g2    = (const float*)d_in[11];
  const float* be2   = (const float*)d_in[12];
  float* out = (float*)d_out;

  float* ws     = (float*)d_ws;
  float* h      = ws;                               // 8M f32
  float* qkv    = h   + (size_t)8 * 1024 * 1024;    // 24M f32
  float* att    = qkv + (size_t)8 * 1024 * 3072;    // 8M f32
  float* tmp    = att + (size_t)8 * 1024 * 1024;    // 8M f32
  float* pooled = tmp + (size_t)8 * 1024 * 1024;    // 8K f32
  float* flin   = pooled + 8 * 1024;                // 8K f32

  const int M = 8 * 1024;
  dim3 blk(256);

  // h = x @ W_in + b_in
  gemm_bias_kernel<<<dim3(1024 / 256, M / 128), blk, 0, stream>>>(x, W_in, b_in, h, M, 1024, 512);

  for (int l = 0; l < 4; ++l) {
    gemm_bias_kernel<<<dim3(3072 / 256, M / 128), blk, 0, stream>>>(
        h, W_qkv + (size_t)l * 1024 * 3072, b_qkv + l * 3072, qkv, M, 3072, 1024);
    attention_kernel<<<dim3(128, 8), blk, 0, stream>>>(qkv, att);
    gemm_bias_kernel<<<dim3(1024 / 256, M / 128), blk, 0, stream>>>(
        att, W_o + (size_t)l * 1024 * 1024, b_o + l * 1024, tmp, M, 1024, 1024);
    ln_kernel<<<dim3(M), blk, 0, stream>>>(tmp, h, g1 + l * 1024, be1 + l * 1024, h, 1024);
  }

  pool_kernel<<<dim3(32), blk, 0, stream>>>(h, pooled);
  smallgemm_kernel<<<dim3(32), blk, 0, stream>>>(pooled, W_out, b_out, flin);
  ln_kernel<<<dim3(8), blk, 0, stream>>>(flin, nullptr, g2, be2, out, 1024);
}
